// NONLocalBlock2D_43301860278976
// MI455X (gfx1250) — matrix operand order repaired and checked by reading, not compiled, not run on hardware
//
#include <hip/hip_runtime.h>
#include <hip/hip_bf16.h>

// ---------------------------------------------------------------------------
// NONLocalBlock2D for MI455X (gfx1250, wave32, WMMA).
// B=8, H=W=64 -> N=4096, C=256, IC=128.
// Fused attention (no 512MB f materialization), bf16 WMMA / f32 accumulate.
// - WMMA B-operands staged transposed in LDS -> ds_load_b128 fragment loads.
// - phi/theta tiles staged with GLOBAL_LOAD_ASYNC_TO_LDS_B128 (ASYNCcnt).
// ---------------------------------------------------------------------------

typedef __attribute__((ext_vector_type(16))) __bf16 v16bf;
typedef __attribute__((ext_vector_type(8)))  float  v8f;

#define NTOK   4096
#define CCH    256
#define ICH    128
#define BNROWS (8 * 4096)

// ---- CDNA5 async copy: global -> LDS, 16B per lane, tracked by ASYNCcnt ----
__device__ __forceinline__ void async_copy_b128(void* lds_ptr,
                                                const void* gptr) {
  // Low 32 bits of a generic LDS pointer are the in-allocation byte offset
  // (aperture check discards the upper bits; see ISA 10.2).
  unsigned lds_off = (unsigned)(unsigned long long)lds_ptr;
  asm volatile("global_load_async_to_lds_b128 %0, %1, off"
               :
               : "v"(lds_off), "v"(gptr)
               : "memory");
}
__device__ __forceinline__ void wait_async0() {
  asm volatile("s_wait_asynccnt 0x0" ::: "memory");
}

// ---- WMMA fragment helpers (layouts per CDNA5 ISA 7.12.2, wave32) ----------

// A matrix 16x32 bf16, from row-major [M][K] LDS tile.
// Lane l(0..15): row M=l, K in {0..7,16..23}; lane l+16: K in {8..15,24..31}.
// Per-lane elements form two 16B-contiguous runs -> 2x ds_load_b128.
__device__ __forceinline__ v16bf load_A(const __bf16* lds, int ld,
                                        int row0, int k0, int lane) {
  const int half = lane >> 4, l = lane & 15;
  v16bf a;
#pragma unroll
  for (int e = 0; e < 16; ++e) {
    int k = ((e >> 3) << 4) + (half << 3) + (e & 7);
    a[e] = lds[(row0 + l) * ld + k0 + k];
  }
  return a;
}

// B matrix 32x16 where B[k][n] = lds[(n0+n)*ld + k0+k]  (transposed [N][K]
// storage). Element e -> K=(e>>3)*16+half*8+(e&7), N=l: two 16B runs per lane.
__device__ __forceinline__ v16bf load_Bt(const __bf16* lds, int ld,
                                         int n0, int k0, int lane) {
  const int half = lane >> 4, l = lane & 15;
  v16bf b;
#pragma unroll
  for (int e = 0; e < 16; ++e) {
    int k = ((e >> 3) << 4) + (half << 3) + (e & 7);
    b[e] = lds[(n0 + l) * ld + k0 + k];
  }
  return b;
}

__device__ __forceinline__ v8f wmma_bf16(v16bf a, v16bf b, v8f c) {
  return __builtin_amdgcn_wmma_f32_16x16x32_bf16(
      false, a, false, b, (short)0, c, false, false);
}

// ---------------------------------------------------------------------------
// Kernel 1: projections. out[p] = bf16(x @ w_p + b_p), p in {theta, phi, g}.
// Block = 256 threads (8 waves), 64 rows x 128 cols per block.
// ---------------------------------------------------------------------------
__global__ __launch_bounds__(256) void proj_kernel(
    const float* __restrict__ x,
    const float* __restrict__ w_th, const float* __restrict__ b_th,
    const float* __restrict__ w_ph, const float* __restrict__ b_ph,
    const float* __restrict__ w_g,  const float* __restrict__ b_g,
    __bf16* __restrict__ theta, __bf16* __restrict__ phi,
    __bf16* __restrict__ g) {
  const int p = blockIdx.y;
  const float* w    = (p == 0) ? w_th : (p == 1) ? w_ph : w_g;
  const float* bias = (p == 0) ? b_th : (p == 1) ? b_ph : b_g;
  __bf16* out       = (p == 0) ? theta : (p == 1) ? phi : g;

  __shared__ __bf16 xs[64][40];     // A tile: 64 rows x 32 K (row-major, pad)
  __shared__ __bf16 wT[128][40];    // B tile TRANSPOSED: [col][K 0..31] (pad)

  const int t = threadIdx.x;
  const int lane = t & 31, wave = t >> 5;
  const int row0 = blockIdx.x * 64;
  const int wr = (wave >> 1) << 4;  // wave row-tile base: 0/16/32/48
  const int wc = (wave & 1) << 6;   // wave col base: 0/64

  v8f acc[4] = {v8f{}, v8f{}, v8f{}, v8f{}};

  for (int k0 = 0; k0 < CCH; k0 += 32) {
    // stage x slice (f32 -> bf16), row-major
    {
      int r = t >> 2, c = (t & 3) * 8;
      const float* src = x + (size_t)(row0 + r) * CCH + k0 + c;
#pragma unroll
      for (int i = 0; i < 8; ++i) xs[r][c + i] = (__bf16)src[i];
    }
    // stage weight slice transposed: global row-major [K][128] -> LDS [col][K]
    {
      int r = t >> 3, c = (t & 7) * 16;  // r = K row 0..31
      const float* src = w + (size_t)(k0 + r) * ICH + c;
#pragma unroll
      for (int i = 0; i < 16; ++i) wT[c + i][r] = (__bf16)src[i];
    }
    __syncthreads();
    v16bf a = load_A(&xs[0][0], 40, wr, 0, lane);
#pragma unroll
    for (int tc = 0; tc < 4; ++tc) {
      v16bf bb = load_Bt(&wT[0][0], 40, wc + tc * 16, 0, lane);
      acc[tc] = wmma_bf16(a, bb, acc[tc]);
    }
    __syncthreads();
  }

  const int half = lane >> 4, l = lane & 15;
#pragma unroll
  for (int tc = 0; tc < 4; ++tc) {
    int col = wc + tc * 16 + l;
    float bv = bias[col];
#pragma unroll
    for (int j = 0; j < 8; ++j) {
      int r = row0 + wr + j + (half ? 8 : 0);
      out[(size_t)r * ICH + col] = (__bf16)(acc[tc][j] + bv);
    }
  }
}

// ---------------------------------------------------------------------------
// Kernel 2: fused attention. For 64 query rows: Y = sum_m (theta . phi^T / N) g
// theta A-fragments hoisted into registers; phi/theta tiles staged via
// async global->LDS DMA; next m-tile prefetched.
// ---------------------------------------------------------------------------
__global__ __launch_bounds__(256) void attn_kernel(
    const __bf16* __restrict__ theta, const __bf16* __restrict__ phi,
    const __bf16* __restrict__ g, __bf16* __restrict__ y) {
  __shared__ __bf16 th_s[64][136];  // theta tile [64 q][128 k]   (A, row-major)
  __shared__ __bf16 ph_s[64][136];  // phi tile   [64 m][128 k]   (B^T natural)
  __shared__ __bf16 gT[128][72];    // g tile TRANSPOSED [128 ic][64 m]
  __shared__ __bf16 s_s[64][72];    // S tile bf16 [64 q][64 m]   (A, row-major)

  const int t = threadIdx.x, lane = t & 31, wave = t >> 5;
  const int half = lane >> 4, l = lane & 15;
  const int q0 = blockIdx.x * 64;
  const size_t base = (size_t)blockIdx.y * NTOK * ICH;

  const int sr = t >> 2, scol = (t & 3) * 32;  // staging coords (64 x 128)

  // stage theta once via async DMA (pure bf16 copy, no VGPR round-trip)
  {
    const __bf16* src = theta + base + (size_t)(q0 + sr) * ICH + scol;
#pragma unroll
    for (int i = 0; i < 4; ++i)
      async_copy_b128(&th_s[sr][scol + i * 8], src + i * 8);
  }
  wait_async0();
  __syncthreads();

  const int wr = (wave >> 1) << 4;   // S/Y row-tile base
  const int wcY = (wave & 1) << 6;   // Y col base
  const int sc0 = (wave & 1) << 5;   // S col base (2 tiles of 16 from here)

  // loop-invariant theta A fragments: 4 x v16bf = 32 VGPRs
  v16bf thf[4];
#pragma unroll
  for (int kk = 0; kk < 4; ++kk)
    thf[kk] = load_A(&th_s[0][0], 136, wr, kk * 32, lane);

  v8f yacc[4] = {v8f{}, v8f{}, v8f{}, v8f{}};
  const float scale = 1.0f / (float)NTOK;

  for (int m0 = 0; m0 < NTOK; m0 += 64) {
    __syncthreads();  // prev-iteration compute reads done before restage
    {
      // phi tile: async DMA straight into LDS
      const __bf16* sp = phi + base + (size_t)(m0 + sr) * ICH + scol;
#pragma unroll
      for (int i = 0; i < 4; ++i)
        async_copy_b128(&ph_s[sr][scol + i * 8], sp + i * 8);
      // g tile: load + transpose into [ic][m] (needed for B-fragment layout)
      __bf16 gtmp[32];
      const uint4* sg = (const uint4*)(g + base + (size_t)(m0 + sr) * ICH + scol);
#pragma unroll
      for (int i = 0; i < 4; ++i) *(uint4*)&gtmp[i * 8] = sg[i];
#pragma unroll
      for (int i = 0; i < 32; ++i) gT[scol + i][sr] = gtmp[i];
      // prefetch next m-tile while this one is consumed (global_prefetch_b8)
      if (m0 + 64 < NTOK) {
        __builtin_prefetch(phi + base + (size_t)(m0 + 64 + sr) * ICH + scol, 0, 1);
        __builtin_prefetch(g + base + (size_t)(m0 + 64 + sr) * ICH + scol, 0, 1);
      }
    }
    wait_async0();
    __syncthreads();

    // S = theta(64x128) @ phi^T : 16 tiles of 16x16, 2 per wave, K=128.
#pragma unroll
    for (int st = 0; st < 2; ++st) {
      int sc = sc0 + st * 16;
      v8f sacc = v8f{};
#pragma unroll
      for (int kk = 0; kk < 4; ++kk) {
        v16bf bb = load_Bt(&ph_s[0][0], 136, sc, kk * 32, lane);
        sacc = wmma_bf16(thf[kk], bb, sacc);
      }
#pragma unroll
      for (int j = 0; j < 8; ++j)
        s_s[wr + j + (half ? 8 : 0)][sc + l] = (__bf16)(sacc[j] * scale);
    }
    __syncthreads();

    // Y += S(64x64) @ g(64x128); per wave 4 tiles, K=64 (2 WMMA steps).
#pragma unroll
    for (int kk = 0; kk < 2; ++kk) {
      v16bf a = load_A(&s_s[0][0], 72, wr, kk * 32, lane);
#pragma unroll
      for (int tc = 0; tc < 4; ++tc) {
        v16bf bb = load_Bt(&gT[0][0], 72, wcY + tc * 16, kk * 32, lane);
        yacc[tc] = wmma_bf16(a, bb, yacc[tc]);
      }
    }
  }

#pragma unroll
  for (int tc = 0; tc < 4; ++tc) {
    int col = wcY + tc * 16 + l;
#pragma unroll
    for (int j = 0; j < 8; ++j) {
      int r = q0 + wr + j + (half ? 8 : 0);
      y[base + (size_t)r * ICH + col] = (__bf16)yacc[tc][j];
    }
  }
}

// ---------------------------------------------------------------------------
// Kernel 3: z = BN(y @ w_W + b_W) + x. Block: 64 rows x 128 cols (grid.y=2).
// ---------------------------------------------------------------------------
__global__ __launch_bounds__(256) void out_kernel(
    const __bf16* __restrict__ y, const float* __restrict__ wW,
    const float* __restrict__ bW, const float* __restrict__ gamma,
    const float* __restrict__ beta, const float* __restrict__ mean,
    const float* __restrict__ var, const float* __restrict__ x,
    float* __restrict__ z) {
  __shared__ __bf16 ys[64][40];     // A tile row-major
  __shared__ __bf16 wT[128][40];    // B tile transposed [col][K]

  const int t = threadIdx.x, lane = t & 31, wave = t >> 5;
  const int row0 = blockIdx.x * 64;
  const int cbase = blockIdx.y * 128;
  const int wr = (wave >> 1) << 4, wc = (wave & 1) << 6;

  v8f acc[4] = {v8f{}, v8f{}, v8f{}, v8f{}};

  for (int k0 = 0; k0 < ICH; k0 += 32) {
    {
      // y tile is a pure bf16 copy: async DMA into LDS
      const __bf16* src = y + (size_t)(row0 + (t >> 2)) * ICH + k0 + (t & 3) * 8;
      async_copy_b128(&ys[t >> 2][(t & 3) * 8], src);
    }
    {
      int r = t >> 3, c = (t & 7) * 16;
      const float* src = wW + (size_t)(k0 + r) * CCH + cbase + c;
#pragma unroll
      for (int i = 0; i < 16; ++i) wT[c + i][r] = (__bf16)src[i];
    }
    wait_async0();
    __syncthreads();
    v16bf a = load_A(&ys[0][0], 40, wr, 0, lane);
#pragma unroll
    for (int tc = 0; tc < 4; ++tc) {
      v16bf bb = load_Bt(&wT[0][0], 40, wc + tc * 16, 0, lane);
      acc[tc] = wmma_bf16(a, bb, acc[tc]);
    }
    __syncthreads();
  }

  const int half = lane >> 4, l = lane & 15;
#pragma unroll
  for (int tc = 0; tc < 4; ++tc) {
    int col = cbase + wc + tc * 16 + l;
    float bv = bW[col], ga = gamma[col], be = beta[col], mu = mean[col];
    float inv = __frsqrt_rn(var[col] + 1e-3f);
#pragma unroll
    for (int j = 0; j < 8; ++j) {
      size_t idx = (size_t)(row0 + wr + j + (half ? 8 : 0)) * CCH + col;
      float wv = acc[tc][j] + bv;
      z[idx] = ga * (wv - mu) * inv + be + x[idx];
    }
  }
}

// ---------------------------------------------------------------------------
extern "C" void kernel_launch(void* const* d_in, const int* in_sizes, int n_in,
                              void* d_out, int out_size, void* d_ws,
                              size_t ws_size, hipStream_t stream) {
  (void)in_sizes; (void)n_in; (void)out_size; (void)ws_size;
  const float* x    = (const float*)d_in[0];
  const float* w_g  = (const float*)d_in[1];
  const float* b_g  = (const float*)d_in[2];
  const float* w_th = (const float*)d_in[3];
  const float* b_th = (const float*)d_in[4];
  const float* w_ph = (const float*)d_in[5];
  const float* b_ph = (const float*)d_in[6];
  const float* w_W  = (const float*)d_in[7];
  const float* b_W  = (const float*)d_in[8];
  const float* gam  = (const float*)d_in[9];
  const float* bet  = (const float*)d_in[10];
  const float* mu   = (const float*)d_in[11];
  const float* var  = (const float*)d_in[12];
  float* z = (float*)d_out;

  char* wsb = (char*)d_ws;
  const size_t seg = (size_t)BNROWS * ICH * sizeof(__bf16);  // 8 MiB
  __bf16* theta = (__bf16*)(wsb);
  __bf16* phi   = (__bf16*)(wsb + seg);
  __bf16* g     = (__bf16*)(wsb + 2 * seg);
  __bf16* y     = (__bf16*)(wsb + 3 * seg);

  proj_kernel<<<dim3(BNROWS / 64, 3, 1), 256, 0, stream>>>(
      x, w_th, b_th, w_ph, b_ph, w_g, b_g, theta, phi, g);
  attn_kernel<<<dim3(NTOK / 64, 8, 1), 256, 0, stream>>>(theta, phi, g, y);
  out_kernel<<<dim3(BNROWS / 64, 2, 1), 256, 0, stream>>>(
      y, w_W, b_W, gam, bet, mu, var, x, z);
}
